// SingleStageFIRFilter_11553462026858
// MI455X (gfx1250) — matrix operand order repaired and checked by reading, not compiled, not run on hardware
//
#include <hip/hip_runtime.h>

#define FILTER_ORDER 24
#define P_FRAME      80
#define IR_LEN       200
#define NFFT         512
#define TAPS         199          // IR_LEN-1
#define WIN          399          // 2*IR_LEN-1
#define KPAD         416          // 13 * 32, K padded for WMMA
#define NCHUNK       13
#define B_BATCH      4
#define T_LEN        48000
#define N_FRAMES     600
#define XWIN         512          // covers 79 + 415 = 494 max offset
#define NCOLS        16           // B padded to full 16 columns (4 used)

typedef _Float16 v16h __attribute__((ext_vector_type(16)));
typedef float    v8f  __attribute__((ext_vector_type(8)));

__global__ __launch_bounds__(160)
void fir_mlsa_wmma(const float* __restrict__ x,
                   const float* __restrict__ mc,
                   float* __restrict__ y)
{
    // ---- LDS staging (~33 KB) ----
    __shared__ float costab[NFFT];            // cos(2*pi*i/512)
    __shared__ float E[2][NFFT];              // exp(logH) for frame n and n+1
    __shared__ float h2[2][IR_LEN];           // impulse responses
    __shared__ __align__(32) _Float16 sB[NCOLS][KPAD]; // 0:HA_hi 1:HB_hi 2:HA_lo 3:HB_lo, 4..15: zero
    __shared__ _Float16 sXhi[XWIN];
    __shared__ _Float16 sXlo[XWIN];
    __shared__ float sOut[5][2][16][16];      // per-wave D tiles (acc1, acc2)

    const int tid  = threadIdx.x;
    const int wave = tid >> 5;
    const int lane = tid & 31;

    const int blk = blockIdx.x;
    const int b   = blk / N_FRAMES;
    const int n   = blk - b * N_FRAMES;
    const int nB  = (n + 1 < N_FRAMES) ? n + 1 : n;  // clamp (h_next of last frame = itself)
    const int t0  = n * P_FRAME;

    const float* mcA = mc + ((size_t)b * N_FRAMES + n)  * (FILTER_ORDER + 1);
    const float* mcB = mc + ((size_t)b * N_FRAMES + nB) * (FILTER_ORDER + 1);

    // Warm the x window (emits global_prefetch_b8 on gfx1250)
    __builtin_prefetch(&x[(size_t)b * T_LEN + t0], 0, 0);

    // ---- cosine table + zero-fill of unused B columns (dword stores) ----
    const float twopi_n = 6.283185307179586f / (float)NFFT;
    for (int i = tid; i < NFFT; i += 160)
        costab[i] = cosf(twopi_n * (float)i);
    {
        unsigned int* zb = (unsigned int*)&sB[4][0];       // cols 4..15: 12*KPAD halves
        const int nd = (NCOLS - 4) * KPAD / 2;             // as dwords
        for (int i = tid; i < nd; i += 160) zb[i] = 0u;
    }
    __syncthreads();

    // ---- spectra: E[s][m] = exp(mc0 + sum_k mc_k * cos(2*pi*k*m/512)) ----
    for (int m = tid; m < 2 * NFFT; m += 160) {
        const int s  = m >> 9;               // 0 = frame n, 1 = frame n+1
        const int mm = m & (NFFT - 1);
        const float* mcs = s ? mcB : mcA;
        float lh = mcs[0];
        int idx = 0;
        #pragma unroll 4
        for (int k = 1; k <= FILTER_ORDER; ++k) {
            idx = (idx + mm) & (NFFT - 1);   // idx = k*mm mod 512
            lh = fmaf(mcs[k], costab[idx], lh);
        }
        E[s][mm] = expf(lh);
    }
    __syncthreads();

    // ---- impulse responses: h[j] = (1/512) * sum_m E[m]*cos(2*pi*j*m/512) ----
    for (int j = tid; j < 2 * IR_LEN; j += 160) {
        const int s  = (j >= IR_LEN) ? 1 : 0;
        const int jj = j - s * IR_LEN;
        const float* Es = E[s];
        float acc = 0.0f;
        int idx = 0;
        #pragma unroll 8
        for (int m = 0; m < NFFT; ++m) {
            acc = fmaf(Es[m], costab[idx], acc);   // idx = jj*m mod 512
            idx = (idx + jj) & (NFFT - 1);
        }
        h2[s][jj] = acc * (1.0f / (float)NFFT);
    }
    __syncthreads();

    // ---- mirrored 399-tap filters, split to f16 hi/lo (Markidis), K padded to 416 ----
    for (int i = tid; i < KPAD; i += 160) {
        int d = i - TAPS; if (d < 0) d = -d;       // hw[i] = h[|i-199|]
        const float va = (i < WIN) ? h2[0][d] : 0.0f;
        const float vb = (i < WIN) ? h2[1][d] : 0.0f;
        const _Float16 ah = (_Float16)va;
        const _Float16 bh = (_Float16)vb;
        sB[0][i] = ah;
        sB[1][i] = bh;
        sB[2][i] = (_Float16)(va - (float)ah);
        sB[3][i] = (_Float16)(vb - (float)bh);
    }
    // ---- x window (with implicit zero padding), split to f16 hi/lo ----
    for (int i = tid; i < XWIN; i += 160) {
        const int gx = t0 + i - TAPS;              // xp[t0+i] = x[t0+i-199]
        const float v = (gx >= 0 && gx < T_LEN) ? x[(size_t)b * T_LEN + gx] : 0.0f;
        const _Float16 hi = (_Float16)v;
        sXhi[i] = hi;
        sXlo[i] = (_Float16)(v - (float)hi);
    }
    __syncthreads();

    // ---- WMMA: Y(80 x {A,B}) = X(80x416) * B(416x4) via 5 M-tiles x 13 K-chunks ----
    const int row   = lane & 15;                    // M within tile / N column
    const int col   = lane & 15;
    const int mbase = wave * 16;
    const int aK0   = (lane < 16) ? 0 : 16;         // A: K half, contiguous order k = 16h + s
    const int bK0   = (lane < 16) ? 0 : 16;         // B: K half, contiguous order k = 16h + s

    v8f acc1 = {};   // x_hi * [HA_hi HB_hi HA_lo HB_lo]
    v8f acc2 = {};   // x_lo * [HA_hi HB_hi  ...  ... ]

    for (int kc = 0; kc < NCHUNK; ++kc) {
        const int kb = kc * 32;
        v16h ahi, alo;
        const int abase = mbase + row + kb + aK0;   // X[p][j] = xp_window[p + j]
        #pragma unroll
        for (int i = 0; i < 16; ++i) {
            const int kl = i;                       // halves 0..15 -> K 0..15 (contiguous)
            ahi[i] = sXhi[abase + kl];
            alo[i] = sXlo[abase + kl];
        }
        // Unconditional aligned 32B load: cols 4..15 are zero-filled, no exec-mask branch
        const v16h bmat = *(const v16h*)&sB[col][kb + bK0];

        acc1 = __builtin_amdgcn_wmma_f32_16x16x32_f16(false, ahi, false, bmat,
                                                      (short)0, acc1, false, false);
        acc2 = __builtin_amdgcn_wmma_f32_16x16x32_f16(false, alo, false, bmat,
                                                      (short)0, acc2, false, false);
    }

    // ---- spill D tiles: VGPR r holds M = r (lanes 0-15) or r+8 (lanes 16-31) ----
    const int rowoff = (lane < 16) ? 0 : 8;
    #pragma unroll
    for (int r = 0; r < 8; ++r) {
        sOut[wave][0][r + rowoff][col] = acc1[r];
        sOut[wave][1][r + rowoff][col] = acc2[r];
    }
    __syncthreads();

    // ---- combine split terms + linear interpolation blend, store y ----
    if (lane < 16) {
        const int p = mbase + lane;                 // 0..79 within frame
        const float YA = sOut[wave][0][lane][0] + sOut[wave][0][lane][2]
                       + sOut[wave][1][lane][0];
        const float YB = sOut[wave][0][lane][1] + sOut[wave][0][lane][3]
                       + sOut[wave][1][lane][1];
        const float w  = (float)p * (1.0f / (float)P_FRAME);
        y[(size_t)b * T_LEN + t0 + p] = (1.0f - w) * YA + w * YB;
    }
}

extern "C" void kernel_launch(void* const* d_in, const int* in_sizes, int n_in,
                              void* d_out, int out_size, void* d_ws, size_t ws_size,
                              hipStream_t stream)
{
    (void)in_sizes; (void)n_in; (void)out_size; (void)d_ws; (void)ws_size;
    const float* x  = (const float*)d_in[0];
    const float* mc = (const float*)d_in[1];
    float* y        = (float*)d_out;

    dim3 grid(B_BATCH * N_FRAMES);   // one workgroup per (batch, frame)
    dim3 block(160);                 // 5 waves of 32: one 16-row M-tile each
    hipLaunchKernelGGL(fir_mlsa_wmma, grid, block, 0, stream, x, mc, y);
}
